// TransformerBlock_O_25486335934506
// MI455X (gfx1250) — matrix operand order repaired and checked by reading, not compiled, not run on hardware
//
#include <hip/hip_runtime.h>

// ---------------------------------------------------------------------------
// CDNA5 (gfx1250, wave32) implementation of the TransformerBlock reference.
// All heavy GEMMs (QKVV, attention, projections, 3x3x3 convs as 27-tap
// implicit GEMM, 1x1x1 conv) run on v_wmma_f32_16x16x32_bf16.
// Activations are kept token-major [B*4096, 256] bf16 so every WMMA fragment
// load is a contiguous 16B dword4; the 192MB L2 keeps operands resident.
// Each wave owns a 16x32 output tile (two accumulators share one A fragment)
// and the K-loop is software-pipelined so global loads overlap the WMMAs.
// ---------------------------------------------------------------------------

#define DEVINL __device__ __forceinline__

typedef unsigned short u16;
typedef unsigned int   u32;
typedef __bf16 v16bf __attribute__((ext_vector_type(16)));
typedef float  v8f   __attribute__((ext_vector_type(8)));

struct U32x8 { uint4 q0, q1; };   // 32 bytes == one 16x(bf16) fragment

static constexpr int Bsz   = 4;
static constexpr int Cch   = 256;
static constexpr int Ntok  = 4096;      // 16*16*16
static constexpr int HEADS = 8;
static constexpr int DH    = 32;
static constexpr int PROJ  = 64;
static constexpr int MTOK  = Bsz * Ntok; // 16384

DEVINL u16 f2bf(float f) {              // round-to-nearest-even f32 -> bf16
  u32 b = __builtin_bit_cast(u32, f);
  b += 0x7FFFu + ((b >> 16) & 1u);
  return (u16)(b >> 16);
}
DEVINL float bf2f(u16 h) { u32 b = (u32)h << 16; return __builtin_bit_cast(float, b); }

DEVINL v8f wmma_bf16(U32x8 a, U32x8 b, v8f c) {
  return __builtin_amdgcn_wmma_f32_16x16x32_bf16(false, __builtin_bit_cast(v16bf, a),
                                                 false, __builtin_bit_cast(v16bf, b),
                                                 (short)0, c, false, false);
}

// A fragment: half-wave 0 holds K {0..7,16..23}, half-wave 1 {8..15,24..31}
DEVINL U32x8 load_afrag(const u16* arow) {
  U32x8 a;
  a.q0 = *(const uint4*)(arow);
  a.q1 = *(const uint4*)(arow + 16);
  return a;
}
// B fragment: half-wave 0 holds K {0..7,16..23}, half-wave 1 {8..15,24..31}
DEVINL U32x8 load_bfrag(const u16* brow) {
  U32x8 b;
  b.q0 = *(const uint4*)(brow);
  b.q1 = *(const uint4*)(brow + 16);
  return b;
}

DEVINL float block_sum_256(float v, float* sm) {
  int t = threadIdx.x;
  sm[t] = v; __syncthreads();
#pragma unroll
  for (int s = 128; s > 0; s >>= 1) { if (t < s) sm[t] += sm[t + s]; __syncthreads(); }
  float r = sm[0]; __syncthreads();
  return r;
}

// ---------------------------------------------------------------------------
// Generic batched WMMA GEMM:  C[m][n] = scale * sum_k A[m][k]*Bt[n][k] + bias
// A,Bt bf16 row-major; C fp32 or bf16. 16x32 tile per wave, 8 waves/block.
// Requires M % 16 == 0, N % 32 == 0, K % 32 == 0.
// Batch index decomposes as (b = batch / hMod, h = batch % hMod).
// ---------------------------------------------------------------------------
struct GemmP {
  const u16* A; const u16* Bt;
  float* Cf; u16* Cb;                       // exactly one non-null
  const float* biasM; const float* biasN; const float* scaleArr;
  long long sAb, sAh, sBb, sBh, sCb, sCh;   // batch strides (elements)
  int lda, ldb, ldc, M, N, K, hMod, scaleMod;
};

__global__ __launch_bounds__(256) void wmma_gemm(GemmP p) {
  const int wave = threadIdx.x >> 5, lane = threadIdx.x & 31;
  const int mt = p.M >> 4, nt2 = p.N >> 5;
  int t = blockIdx.x * 8 + wave;
  if (t >= mt * nt2) return;                     // wave-uniform guard
  const int tm = t % mt, tn = t / mt;            // waves in a block share tn
  const int batch = blockIdx.y;
  const int bb = batch / p.hMod, hh = batch % p.hMod;
  const u16* A  = p.A  + (long long)bb * p.sAb + (long long)hh * p.sAh;
  const u16* Bt = p.Bt + (long long)bb * p.sBb + (long long)hh * p.sBh;
  const int half = lane >> 4, lr = lane & 15;
  const u16* arow  = A  + (long long)(tm * 16 + lr) * p.lda + (half << 3);
  const u16* brow0 = Bt + (long long)(tn * 32 + lr) * p.ldb + (half << 3);
  const u16* brow1 = brow0 + (long long)16 * p.ldb;
  v8f acc0 = {0.f,0.f,0.f,0.f,0.f,0.f,0.f,0.f};
  v8f acc1 = {0.f,0.f,0.f,0.f,0.f,0.f,0.f,0.f};
  // software pipeline: fragments for step k are in flight while step k-32 MACs
  U32x8 au = load_afrag(arow), bu0 = load_bfrag(brow0), bu1 = load_bfrag(brow1);
  for (int k0 = 32; k0 < p.K; k0 += 32) {
    if (k0 + 32 < p.K) {                         // global_prefetch_b8
      __builtin_prefetch(arow + k0 + 32, 0, 1);
      __builtin_prefetch(brow0 + k0 + 32, 0, 1);
    }
    U32x8 an  = load_afrag(arow + k0);
    U32x8 bn0 = load_bfrag(brow0 + k0);
    U32x8 bn1 = load_bfrag(brow1 + k0);
    acc0 = wmma_bf16(au, bu0, acc0);
    acc1 = wmma_bf16(au, bu1, acc1);
    au = an; bu0 = bn0; bu1 = bn1;
  }
  acc0 = wmma_bf16(au, bu0, acc0);
  acc1 = wmma_bf16(au, bu1, acc1);

  const float scale = p.scaleArr ? p.scaleArr[batch % p.scaleMod] : 1.0f;
  const int gn0 = tn * 32 + lr, gn1 = gn0 + 16;
  const float bN0 = p.biasN ? p.biasN[gn0] : 0.0f;
  const float bN1 = p.biasN ? p.biasN[gn1] : 0.0f;
  const long long cbase = (long long)bb * p.sCb + (long long)hh * p.sCh;
#pragma unroll
  for (int r = 0; r < 8; ++r) {
    const int gm = tm * 16 + (half << 3) + r;    // C layout: M = r + 8*halfwave
    const float bM = p.biasM ? p.biasM[gm] : 0.0f;
    const float v0 = acc0[r] * scale + bN0 + bM;
    const float v1 = acc1[r] * scale + bN1 + bM;
    const long long row = cbase + (long long)gm * p.ldc;
    if (p.Cf) { p.Cf[row + gn0] = v0;      p.Cf[row + gn1] = v1; }
    else      { p.Cb[row + gn0] = f2bf(v0); p.Cb[row + gn1] = f2bf(v1); }
  }
}

// ---------------------------------------------------------------------------
// 3x3x3 conv, C=256->256, pad 1, as 27-tap implicit GEMM in token-major layout.
// In  : [B][4096][256] bf16 (token n = (s0*16+s1)*16+s2)
// Wp  : [27][256(o)][256(i)] bf16 (pre-packed)
// Out : [B][4096][256] fp32 (+bias)
// Wave tile: 16 pixels (s2) x 32 output channels; 8 waves cover all 256 o.
// grid = (B*256, 1), block 256. K-loop per tap is software-pipelined.
// ---------------------------------------------------------------------------
__global__ __launch_bounds__(256) void wmma_conv3(const u16* __restrict__ In,
                                                  const u16* __restrict__ Wp,
                                                  const float* __restrict__ bias,
                                                  float* __restrict__ Out) {
  const int wave = threadIdx.x >> 5, lane = threadIdx.x & 31;
  const int prow = blockIdx.x;                       // b*256 + s0*16 + s1
  const int b = prow >> 8, s0 = (prow >> 4) & 15, s1 = prow & 15;
  const int half = lane >> 4, lr = lane & 15;
  const u16* inb = In + ((long long)b * Ntok) * Cch;
  v8f acc0 = {0.f,0.f,0.f,0.f,0.f,0.f,0.f,0.f};
  v8f acc1 = {0.f,0.f,0.f,0.f,0.f,0.f,0.f,0.f};
  for (int t0 = -1; t0 <= 1; ++t0) {
    const int z = s0 + t0; if ((unsigned)z > 15u) continue;
    for (int t1 = -1; t1 <= 1; ++t1) {
      const int y = s1 + t1; if ((unsigned)y > 15u) continue;
      const int rowbase = (z * 16 + y) * 16;
      for (int t2 = -1; t2 <= 1; ++t2) {
        const int x2 = lr + t2;                      // shifted input pixel (per-lane)
        const bool ok = (unsigned)x2 < 16u;          // zero-pad rows at s2 edges
        const u16* arow = inb + (long long)(rowbase + (ok ? x2 : 0)) * Cch + (half << 3);
        const int tap = (t0 + 1) * 9 + (t1 + 1) * 3 + (t2 + 1);
        const u16* brow0 = Wp + ((long long)tap * Cch + wave * 32 + lr) * Cch + (half << 3);
        const u16* brow1 = brow0 + 16 * Cch;
        U32x8 au = {}, bu0, bu1;
        if (ok) au = load_afrag(arow);
        bu0 = load_bfrag(brow0);
        bu1 = load_bfrag(brow1);
#pragma unroll
        for (int k0 = 32; k0 < Cch; k0 += 32) {
          U32x8 an = {}, bn0, bn1;
          if (ok) an = load_afrag(arow + k0);
          bn0 = load_bfrag(brow0 + k0);
          bn1 = load_bfrag(brow1 + k0);
          acc0 = wmma_bf16(au, bu0, acc0);
          acc1 = wmma_bf16(au, bu1, acc1);
          au = an; bu0 = bn0; bu1 = bn1;
        }
        acc0 = wmma_bf16(au, bu0, acc0);
        acc1 = wmma_bf16(au, bu1, acc1);
      }
    }
  }
  const int o0 = wave * 32 + lr, o1 = o0 + 16;
  const float bb0 = bias[o0], bb1 = bias[o1];
  float* crow = Out + ((long long)(b * Ntok + (s0 * 16 + s1) * 16)) * Cch;
#pragma unroll
  for (int r = 0; r < 8; ++r) {
    const int s2 = (half << 3) + r;
    crow[(long long)s2 * Cch + o0] = acc0[r] + bb0;
    crow[(long long)s2 * Cch + o1] = acc1[r] + bb1;
  }
}

// ------------------------- small helper kernels ----------------------------

// LayerNorm over C; x is [B][C][4096]; writes token-major bf16 [B*4096][256]
__global__ __launch_bounds__(256) void ln_kernel(const float* __restrict__ x,
                                                 const float* __restrict__ s,
                                                 const float* __restrict__ bias,
                                                 u16* __restrict__ out) {
  __shared__ float sm[256];
  const int r = blockIdx.x;                 // b*4096 + n
  const int b = r >> 12, n = r & 4095, c = threadIdx.x;
  const float v = x[(((long long)b * Cch + c) << 12) + n];
  const float mean = block_sum_256(v, sm) * (1.0f / 256.0f);
  const float d = v - mean;
  const float var = block_sum_256(d * d, sm) * (1.0f / 256.0f);
  const float y = d * __frsqrt_rn(var + 1e-5f) * s[c] + bias[c];
  out[(long long)r * Cch + c] = f2bf(y);
}

// Per-(b,h,d) row over n=4096: l2-normalize q,k along n; emit the layouts the
// attention GEMMs need (q/k transposed to [d][n], q also [n][d], raw k, raw v_sa)
__global__ __launch_bounds__(256) void prep_qkvv(const u16* __restrict__ qkvv,
                                                 u16* qn_dn, u16* qn_nd,
                                                 u16* kn_dn, u16* k_dn, u16* vsa_dn) {
  __shared__ float sm[256];
  const int row = blockIdx.x;               // (b*8+h)*32 + d
  const int d = row & 31, h = (row >> 5) & 7, b = row >> 8;
  const int qcol = h * DH + d, kcol = Cch + h * DH + d, vcol = 3 * Cch + h * DH + d;
  const long long base = ((long long)b * Ntok) * 1024;
  float qv[16], kv[16], vv[16];
  float q2 = 0.f, k2 = 0.f;
#pragma unroll
  for (int i = 0; i < 16; ++i) {
    const long long rowoff = base + (long long)(threadIdx.x + i * 256) * 1024;
    qv[i] = bf2f(qkvv[rowoff + qcol]);
    kv[i] = bf2f(qkvv[rowoff + kcol]);
    vv[i] = bf2f(qkvv[rowoff + vcol]);
    q2 += qv[i] * qv[i]; k2 += kv[i] * kv[i];
  }
  q2 = block_sum_256(q2, sm);
  k2 = block_sum_256(k2, sm);
  const float rq = 1.0f / fmaxf(__fsqrt_rn(q2), 1e-12f);
  const float rk = 1.0f / fmaxf(__fsqrt_rn(k2), 1e-12f);
  const long long dn = (long long)row * Ntok;
  const long long ndb = ((long long)(b * HEADS + h)) * Ntok;
#pragma unroll
  for (int i = 0; i < 16; ++i) {
    const int n = threadIdx.x + i * 256;
    qn_dn[dn + n]  = f2bf(qv[i] * rq);
    qn_nd[(ndb + n) * DH + d] = f2bf(qv[i] * rq);
    kn_dn[dn + n]  = f2bf(kv[i] * rk);
    k_dn [dn + n]  = f2bf(kv[i]);
    vsa_dn[dn + n] = f2bf(vv[i]);
  }
}

// softmax over rows of 32*VPT (wave32 shuffle reduction), bf16 out
template<int VPT>
__global__ __launch_bounds__(256) void softmax_rows(const float* __restrict__ in,
                                                    u16* __restrict__ out, int rows) {
  const int wid = (blockIdx.x * blockDim.x + threadIdx.x) >> 5;
  if (wid >= rows) return;
  const int lane = threadIdx.x & 31;
  const float* r = in + (long long)wid * (32 * VPT);
  float v[VPT]; float mx = -3.4e38f;
#pragma unroll
  for (int i = 0; i < VPT; ++i) { v[i] = r[lane + i * 32]; mx = fmaxf(mx, v[i]); }
#pragma unroll
  for (int m = 16; m; m >>= 1) mx = fmaxf(mx, __shfl_xor(mx, m, 32));
  float s = 0.f;
#pragma unroll
  for (int i = 0; i < VPT; ++i) { v[i] = __expf(v[i] - mx); s += v[i]; }
#pragma unroll
  for (int m = 16; m; m >>= 1) s += __shfl_xor(s, m, 32);
  const float inv = 1.0f / s;
  u16* o = out + (long long)wid * (32 * VPT);
#pragma unroll
  for (int i = 0; i < VPT; ++i) o[lane + i * 32] = f2bf(v[i] * inv);
}

// faithful permute(0,3,1,2).reshape(B,N,C) of x_sa [b][h][n][d]
__global__ __launch_bounds__(256) void permute_xsa(const float* __restrict__ xsa,
                                                   u16* __restrict__ dst) {
  long long i = (long long)blockIdx.x * 256 + threadIdx.x;  // B*H*N*DH elems
  const int d = i & 31; long long r = i >> 5;
  const int n = r & 4095; r >>= 12;
  const int h = r & 7; const int b = (int)(r >> 3);
  const int row = (d * HEADS + h) * 16 + (n >> 8);
  const int col = n & 255;
  dst[(((long long)b * Ntok + row) << 8) + col] = f2bf(xsa[i]);
}

__global__ __launch_bounds__(256) void convert_bf(const float* __restrict__ src,
                                                  u16* __restrict__ dst, long long n) {
  long long i = (long long)blockIdx.x * 256 + threadIdx.x;
  if (i < n) dst[i] = f2bf(src[i]);
}

// dst[c*R + r] = bf16(src[r*C + c])
__global__ __launch_bounds__(256) void transpose_bf(const float* __restrict__ src,
                                                    u16* __restrict__ dst, int R, int C) {
  long long i = (long long)blockIdx.x * 256 + threadIdx.x;
  if (i >= (long long)R * C) return;
  const int r = (int)(i / C), c = (int)(i % C);
  dst[(long long)c * R + r] = f2bf(src[i]);
}

// conv weights [O][I][3][3][3] -> [tap][o][i] bf16
__global__ __launch_bounds__(256) void pack_conv_w(const float* __restrict__ w,
                                                   u16* __restrict__ wp) {
  long long i = (long long)blockIdx.x * 256 + threadIdx.x;   // 27*256*256
  if (i >= 27LL * 65536) return;
  const int tap = (int)(i >> 16);
  const int o = (int)((i >> 8) & 255), ci = (int)(i & 255);
  wp[i] = f2bf(w[(long long)(o * Cch + ci) * 27 + tap]);
}

// attn = tokens + gamma*epa; emit fp32 token-major, bf16 token-major, fp32 NC-major
__global__ __launch_bounds__(256) void epa_combine(const float* __restrict__ x,
                                                   const float* __restrict__ gamma,
                                                   const float* __restrict__ epa,
                                                   float* __restrict__ attn_t,
                                                   u16* __restrict__ attn_bf,
                                                   float* __restrict__ attn_skip) {
  const int r = blockIdx.x, c = threadIdx.x;     // r = b*4096+n
  const int b = r >> 12, n = r & 4095;
  const float tok = x[(((long long)b * Cch + c) << 12) + n];
  const float a = tok + gamma[c] * epa[(long long)r * Cch + c];
  attn_t [(long long)r * Cch + c] = a;
  attn_bf[(long long)r * Cch + c] = f2bf(a);
  attn_skip[(((long long)b * Cch + c) << 12) + n] = a;
}

// per-channel mean / rstd over 16384 token rows (token-major input)
__global__ __launch_bounds__(256) void bn_stats(const float* __restrict__ in,
                                                float* __restrict__ mean,
                                                float* __restrict__ rstd) {
  __shared__ float sm[256];
  const int c = blockIdx.x;
  float s = 0.f, ss = 0.f;
  for (int r = threadIdx.x; r < MTOK; r += 256) {
    const float v = in[(long long)r * Cch + c];
    s += v; ss += v * v;
  }
  s  = block_sum_256(s,  sm);
  ss = block_sum_256(ss, sm);
  if (threadIdx.x == 0) {
    const float m = s * (1.0f / MTOK);
    const float var = ss * (1.0f / MTOK) - m * m;
    mean[c] = m;
    rstd[c] = __frsqrt_rn(var + 1e-5f);
  }
}

// y = lrelu( bn(in) [+ res] ) -> bf16 (token-major)
__global__ __launch_bounds__(256) void bn_apply(const float* __restrict__ in,
                                                const float* __restrict__ s,
                                                const float* __restrict__ b,
                                                const float* __restrict__ mean,
                                                const float* __restrict__ rstd,
                                                const float* __restrict__ res,
                                                u16* __restrict__ out) {
  const long long i = (long long)blockIdx.x * 256 + threadIdx.x;
  const int c = (int)(i & 255);
  float y = (in[i] - mean[c]) * rstd[c] * s[c] + b[c];
  if (res) y += res[i];
  y = y > 0.f ? y : 0.01f * y;
  out[i] = f2bf(y);
}

// out[b][c][n] = attn_skip[b][c][n] + out8_t[b][n][c]
__global__ __launch_bounds__(256) void final_add(const float* __restrict__ skip,
                                                 const float* __restrict__ out8_t,
                                                 float* __restrict__ out) {
  const long long i = (long long)blockIdx.x * 256 + threadIdx.x;  // B*C*N
  const int n = (int)(i & 4095);
  const int c = (int)((i >> 12) & 255);
  const int b = (int)(i >> 20);
  out[i] = skip[i] + out8_t[(((long long)b << 12) + n) * Cch + c];
}

// ---------------------------------------------------------------------------
extern "C" void kernel_launch(void* const* d_in, const int* in_sizes, int n_in,
                              void* d_out, int out_size, void* d_ws, size_t ws_size,
                              hipStream_t stream) {
  const float* x        = (const float*)d_in[0];
  const float* ln_s     = (const float*)d_in[1];
  const float* ln_b     = (const float*)d_in[2];
  const float* gamma    = (const float*)d_in[3];
  const float* Wqkvv    = (const float*)d_in[4];
  const float* temp     = (const float*)d_in[5];
  const float* temp2    = (const float*)d_in[6];
  const float* E_w      = (const float*)d_in[7];
  const float* E_b      = (const float*)d_in[8];
  const float* outp_w   = (const float*)d_in[9];
  const float* outp_b   = (const float*)d_in[10];
  const float* outp2_w  = (const float*)d_in[11];
  const float* outp2_b  = (const float*)d_in[12];
  const float* conv1_w  = (const float*)d_in[13];
  const float* conv1_b  = (const float*)d_in[14];
  const float* bn1_s    = (const float*)d_in[15];
  const float* bn1_b    = (const float*)d_in[16];
  const float* conv2_w  = (const float*)d_in[17];
  const float* conv2_b  = (const float*)d_in[18];
  const float* bn2_s    = (const float*)d_in[19];
  const float* bn2_b    = (const float*)d_in[20];
  const float* conv8_w  = (const float*)d_in[21];
  const float* conv8_b  = (const float*)d_in[22];
  float* out = (float*)d_out;

  size_t off = 0;
  auto alloc = [&](size_t bytes) -> void* {
    void* p = (char*)d_ws + off;
    off += (bytes + 255) & ~(size_t)255;
    return p;
  };
  const long long TOKC = (long long)MTOK * Cch;             // 4,194,304
  u16*   tok_ln   = (u16*)  alloc(TOKC * 2);
  u16*   wqkvv_t  = (u16*)  alloc((size_t)1024 * 256 * 2);
  u16*   e_t      = (u16*)  alloc((size_t)PROJ * Ntok * 2);
  u16*   outp_t   = (u16*)  alloc((size_t)128 * 256 * 2);
  u16*   outp2_t  = (u16*)  alloc((size_t)128 * 256 * 2);
  u16*   w8_t     = (u16*)  alloc((size_t)256 * 256 * 2);
  u16*   wp1      = (u16*)  alloc((size_t)27 * 65536 * 2);
  u16*   wp2      = (u16*)  alloc((size_t)27 * 65536 * 2);
  u16*   qkvv_bf  = (u16*)  alloc((size_t)MTOK * 1024 * 2);
  u16*   qn_dn    = (u16*)  alloc(TOKC * 2);
  u16*   qn_nd    = (u16*)  alloc(TOKC * 2);
  u16*   kn_dn    = (u16*)  alloc(TOKC * 2);
  u16*   k_dn     = (u16*)  alloc(TOKC * 2);
  u16*   vsa_dn   = (u16*)  alloc(TOKC * 2);
  float* ca_log   = (float*)alloc((size_t)32 * 32 * 32 * 4);
  u16*   ca_soft  = (u16*)  alloc((size_t)32 * 32 * 32 * 2);
  u16*   kprojT   = (u16*)  alloc((size_t)32 * PROJ * DH * 2);
  u16*   vsaproj  = (u16*)  alloc((size_t)32 * DH * PROJ * 2);
  float* sa_log   = (float*)alloc((size_t)32 * Ntok * PROJ * 4);
  u16*   sa_soft  = (u16*)  alloc((size_t)32 * Ntok * PROJ * 2);
  float* xca_t    = (float*)alloc(TOKC * 4);
  u16*   xca_bf   = (u16*)  alloc(TOKC * 2);
  float* xsa      = (float*)alloc(TOKC * 4);
  u16*   xsa_rs   = (u16*)  alloc(TOKC * 2);
  float* epa      = (float*)alloc(TOKC * 4);
  float* attn_t   = (float*)alloc(TOKC * 4);
  u16*   attn_bf  = (u16*)  alloc(TOKC * 2);
  float* attn_sk  = (float*)alloc(TOKC * 4);
  float* cbuf     = (float*)alloc(TOKC * 4);      // conv1 out, reused for conv2 out
  u16*   act1_bf  = (u16*)  alloc(TOKC * 2);
  u16*   act2_bf  = (u16*)  alloc(TOKC * 2);
  float* out8_t   = (float*)alloc(TOKC * 4);
  float* bnm1     = (float*)alloc(256 * 4);
  float* bnr1     = (float*)alloc(256 * 4);
  float* bnm2     = (float*)alloc(256 * 4);
  float* bnr2     = (float*)alloc(256 * 4);
  (void)ws_size; (void)in_sizes; (void)n_in; (void)out_size;

  auto gemm = [&](GemmP p, int batches) {
    const int tiles = (p.M / 16) * (p.N / 32);
    dim3 grid((tiles + 7) / 8, batches);
    wmma_gemm<<<grid, 256, 0, stream>>>(p);
  };

  // ---- weight pre-packing (bf16) ----
  transpose_bf<<<(256 * 1024 + 255) / 256, 256, 0, stream>>>(Wqkvv, wqkvv_t, 256, 1024);
  transpose_bf<<<(4096 * 64 + 255) / 256, 256, 0, stream>>>(E_w, e_t, 4096, 64);
  transpose_bf<<<(256 * 128 + 255) / 256, 256, 0, stream>>>(outp_w, outp_t, 256, 128);
  transpose_bf<<<(256 * 128 + 255) / 256, 256, 0, stream>>>(outp2_w, outp2_t, 256, 128);
  convert_bf<<<(65536 + 255) / 256, 256, 0, stream>>>(conv8_w, w8_t, 65536);
  pack_conv_w<<<(27 * 65536 + 255) / 256, 256, 0, stream>>>(conv1_w, wp1);
  pack_conv_w<<<(27 * 65536 + 255) / 256, 256, 0, stream>>>(conv2_w, wp2);

  // ---- LayerNorm -> bf16 tokens ----
  ln_kernel<<<MTOK, 256, 0, stream>>>(x, ln_s, ln_b, tok_ln);

  // ---- QKVV GEMM: [16384x256]x[256x1024] -> bf16 token-major ----
  { GemmP p{}; p.A = tok_ln; p.Bt = wqkvv_t; p.Cb = qkvv_bf;
    p.lda = 256; p.ldb = 256; p.ldc = 1024; p.M = MTOK; p.N = 1024; p.K = 256;
    p.hMod = 1; p.scaleMod = 1; gemm(p, 1); }

  // ---- l2norm + layout prep for q,k,v_sa ----
  prep_qkvv<<<Bsz * HEADS * DH, 256, 0, stream>>>(qkvv_bf, qn_dn, qn_nd, kn_dn, k_dn, vsa_dn);

  // ---- channel-attn logits [d x e] = qn.kn^T * temp[h]  (batched over b,h) ----
  { GemmP p{}; p.A = qn_dn; p.Bt = kn_dn; p.Cf = ca_log;
    p.lda = 4096; p.ldb = 4096; p.ldc = 32; p.M = 32; p.N = 32; p.K = 4096;
    p.sAh = 131072; p.sAb = 1048576; p.sBh = 131072; p.sBb = 1048576;
    p.sCh = 1024; p.sCb = 8192; p.hMod = 8;
    p.scaleArr = temp; p.scaleMod = 8; gemm(p, 32); }

  // ---- k_proj^T [p x d] = E^T . k^T + E_b (per-M bias) -> bf16 ----
  { GemmP p{}; p.A = e_t; p.Bt = k_dn; p.Cb = kprojT;
    p.lda = 4096; p.ldb = 4096; p.ldc = 32; p.M = PROJ; p.N = 32; p.K = 4096;
    p.sBh = 131072; p.sBb = 1048576; p.sCh = 2048; p.sCb = 16384;
    p.biasM = E_b; p.hMod = 8; p.scaleMod = 1; gemm(p, 32); }

  // ---- v_sa_proj [d x p] = v_sa . E + E_b -> bf16 ----
  { GemmP p{}; p.A = vsa_dn; p.Bt = e_t; p.Cb = vsaproj;
    p.lda = 4096; p.ldb = 4096; p.ldc = 64; p.M = 32; p.N = PROJ; p.K = 4096;
    p.sAh = 131072; p.sAb = 1048576; p.sCh = 2048; p.sCb = 16384;
    p.biasN = E_b; p.hMod = 8; p.scaleMod = 1; gemm(p, 32); }

  // ---- channel-attn softmax (rows of 32) -> bf16 ----
  softmax_rows<1><<<(Bsz * HEADS * 32) / 8, 256, 0, stream>>>(ca_log, ca_soft, Bsz * HEADS * 32);

  // ---- spatial-attn logits [n x p] = qn^T . k_proj * temp2[h] ----
  { GemmP p{}; p.A = qn_nd; p.Bt = kprojT; p.Cf = sa_log;
    p.lda = 32; p.ldb = 32; p.ldc = 64; p.M = Ntok; p.N = PROJ; p.K = 32;
    p.sAh = 131072; p.sAb = 1048576; p.sBh = 2048; p.sBb = 16384;
    p.sCh = 262144; p.sCb = 2097152; p.hMod = 8;
    p.scaleArr = temp2; p.scaleMod = 8; gemm(p, 32); }

  // ---- spatial-attn softmax (rows of 64) -> bf16 ----
  softmax_rows<2><<<(Bsz * HEADS * Ntok) / 8, 256, 0, stream>>>(sa_log, sa_soft, Bsz * HEADS * Ntok);

  // ---- x_ca^T [n x d] = v_ca^T . attn_ca^T  (writes straight into token layout) ----
  { GemmP p{}; p.A = qkvv_bf + 512; p.Bt = ca_soft; p.Cf = xca_t;
    p.lda = 1024; p.ldb = 32; p.ldc = 256; p.M = Ntok; p.N = 32; p.K = 32;
    p.sAh = 32; p.sAb = (long long)Ntok * 1024; p.sBh = 1024; p.sBb = 8192;
    p.sCh = 32; p.sCb = 1048576; p.hMod = 8; p.scaleMod = 1; gemm(p, 32); }

  // ---- x_sa [n x d] = attn_sa . v_sa_proj^T ----
  { GemmP p{}; p.A = sa_soft; p.Bt = vsaproj; p.Cf = xsa;
    p.lda = 64; p.ldb = 64; p.ldc = 32; p.M = Ntok; p.N = 32; p.K = PROJ;
    p.sAh = 262144; p.sAb = 2097152; p.sBh = 2048; p.sBb = 16384;
    p.sCh = 131072; p.sCb = 1048576; p.hMod = 8; p.scaleMod = 1; gemm(p, 32); }

  // ---- faithful permutes / converts ----
  permute_xsa<<<(int)(TOKC / 256), 256, 0, stream>>>(xsa, xsa_rs);
  convert_bf<<<(int)(TOKC / 256), 256, 0, stream>>>(xca_t, xca_bf, TOKC);

  // ---- output projections: epa = concat(x_sa@outp + b, x_ca@outp2 + b2) ----
  { GemmP p{}; p.A = xsa_rs; p.Bt = outp_t; p.Cf = epa;
    p.lda = 256; p.ldb = 256; p.ldc = 256; p.M = MTOK; p.N = 128; p.K = 256;
    p.biasN = outp_b; p.hMod = 1; p.scaleMod = 1; gemm(p, 1); }
  { GemmP p{}; p.A = xca_bf; p.Bt = outp2_t; p.Cf = epa + 128;
    p.lda = 256; p.ldb = 256; p.ldc = 256; p.M = MTOK; p.N = 128; p.K = 256;
    p.biasN = outp2_b; p.hMod = 1; p.scaleMod = 1; gemm(p, 1); }

  // ---- residual: attn = tokens + gamma*epa ----
  epa_combine<<<MTOK, 256, 0, stream>>>(x, gamma, epa, attn_t, attn_bf, attn_sk);

  // ---- conv1 (3x3x3 implicit GEMM) -> BN -> lrelu -> bf16 ----
  wmma_conv3<<<dim3(Bsz * 256, 1), 256, 0, stream>>>(attn_bf, wp1, conv1_b, cbuf);
  bn_stats<<<256, 256, 0, stream>>>(cbuf, bnm1, bnr1);
  bn_apply<<<(int)(TOKC / 256), 256, 0, stream>>>(cbuf, bn1_s, bn1_b, bnm1, bnr1, nullptr, act1_bf);

  // ---- conv2 -> BN -> +skip -> lrelu -> bf16 ----
  wmma_conv3<<<dim3(Bsz * 256, 1), 256, 0, stream>>>(act1_bf, wp2, conv2_b, cbuf);
  bn_stats<<<256, 256, 0, stream>>>(cbuf, bnm2, bnr2);
  bn_apply<<<(int)(TOKC / 256), 256, 0, stream>>>(cbuf, bn2_s, bn2_b, bnm2, bnr2, attn_t, act2_bf);

  // ---- conv8 (1x1x1 GEMM) ----
  { GemmP p{}; p.A = act2_bf; p.Bt = w8_t; p.Cf = out8_t;
    p.lda = 256; p.ldb = 256; p.ldc = 256; p.M = MTOK; p.N = 256; p.K = 256;
    p.biasN = conv8_b; p.hMod = 1; p.scaleMod = 1; gemm(p, 1); }

  // ---- final: out = attn_skip + out8 (transpose back to [B,C,16,16,16]) ----
  final_add<<<(int)(TOKC / 256), 256, 0, stream>>>(attn_sk, out8_t, out);
}
